// MeshGraphNet_57234734186524
// MI455X (gfx1250) — matrix operand order repaired and checked by reading, not compiled, not run on hardware
//
#include <hip/hip_runtime.h>
#include <hip/hip_bf16.h>

// MeshGraphNet forward for MI455X (gfx1250, wave32, WMMA).
// All GEMMs run on v_wmma_f32_16x16x32_f16: one wave owns a 16-row tile and
// the full 128-wide output (8 column tiles), K-looping 32 at a time.
// Weights are packed once per launch into f16 W^T [Nout x Kpad] so B-fragments
// are two contiguous 16B loads per lane. Activations are staged to LDS as f16
// with padded strides so A-fragments are aligned ds_load_b128 pairs.
// Bandwidth optimizations (HBM-roofline bound at ~23.3 TB/s):
//  - f16 shadow of nf: halves the two random 128-wide gathers per edge.
//  - fp32 ef/nf tiles parked in LDS between stage and residual: no global
//    re-read in the epilogue (store-only residual).

typedef _Float16 h16;
typedef __attribute__((ext_vector_type(16))) _Float16 v16h;
typedef __attribute__((ext_vector_type(8)))  float    v8f;

#define NNODES 50000
#define NEDGES 600000
#define HDIM   128
#define STEPS  15
#define LN_EPS 1e-5f

union FragU { v16h v; uint4 q[2]; };

// A fragment (16x32 f16) from row-major LDS tile. Contiguous K (matches B):
// lanes 0-15: M=lane, K = 0..15; lanes 16-31: K = 16..31.
__device__ __forceinline__ v16h ldsAFrag(const h16* base, int strideH, int lane, int kbase) {
    const int row = lane & 15, hi = lane >> 4;
    const h16* p = base + row * strideH + kbase + hi * 16;
    FragU f;
    f.q[0] = *(const uint4*)(p);
    f.q[1] = *(const uint4*)(p + 8);
    return f.v;
}

// B fragment (32x16 f16) from packed W^T [Npad x Kpad] (f16, row-major):
// lane holds column n = nbase + (lane&15), K = kbase + (lane>>4)*16 + 0..15.
__device__ __forceinline__ v16h glbBFrag(const h16* Wt, int Kpad, int nbase, int kbase, int lane) {
    const int lo = lane & 15, hi = lane >> 4;
    const h16* p = Wt + (size_t)(nbase + lo) * Kpad + kbase + hi * 16;
    FragU f;
    f.q[0] = ((const uint4*)p)[0];
    f.q[1] = ((const uint4*)p)[1];
    return f.v;
}

__device__ __forceinline__ v8f wmma16(v16h a, v16h b, v8f c) {
    return __builtin_amdgcn_wmma_f32_16x16x32_f16(false, a, false, b, (short)0, c, false, false);
}

__device__ __forceinline__ void zeroAcc(v8f acc[8]) {
#pragma unroll
    for (int ct = 0; ct < 8; ++ct)
#pragma unroll
        for (int r = 0; r < 8; ++r) acc[ct][r] = 0.0f;
}

// C tile [16 rows x 128 cols] = 8 accumulators; C layout: M = r + 8*(lane>>4), N = ct*16 + (lane&15)
__device__ __forceinline__ void waveGemm(const h16* aLds, int strideA,
                                         const h16* __restrict__ Wt, int Kpad,
                                         v8f acc[8], int lane) {
    for (int kc = 0; kc < Kpad; kc += 32) {
        v16h a = ldsAFrag(aLds, strideA, lane, kc);
#pragma unroll
        for (int ct = 0; ct < 8; ++ct) {
            v16h b = glbBFrag(Wt, Kpad, ct * 16, kc, lane);
            acc[ct] = wmma16(a, b, acc[ct]);
        }
    }
}

__device__ __forceinline__ void biasAct(v8f acc[8], const float* __restrict__ b, int lane, bool relu) {
    const int lo = lane & 15;
#pragma unroll
    for (int ct = 0; ct < 8; ++ct) {
        const float bb = b[ct * 16 + lo];
#pragma unroll
        for (int r = 0; r < 8; ++r) {
            float v = acc[ct][r] + bb;
            acc[ct][r] = relu ? fmaxf(v, 0.0f) : v;
        }
    }
}

__device__ __forceinline__ void storeHidden(const v8f acc[8], h16* hLds, int strideH, int lane) {
    const int lo = lane & 15, hi = lane >> 4;
#pragma unroll
    for (int ct = 0; ct < 8; ++ct)
#pragma unroll
        for (int r = 0; r < 8; ++r)
            hLds[(r + 8 * hi) * strideH + ct * 16 + lo] = (h16)acc[ct][r];
}

__device__ __forceinline__ void storeOutF(const v8f acc[8], float* fLds, int strideF, int lane) {
    const int lo = lane & 15, hi = lane >> 4;
#pragma unroll
    for (int ct = 0; ct < 8; ++ct)
#pragma unroll
        for (int r = 0; r < 8; ++r)
            fLds[(r + 8 * hi) * strideF + ct * 16 + lo] = acc[ct][r];
}

// In-register LayerNorm over the 128-wide row: per-reg partials + shfl_xor over
// the 16-lane half group (rows live in fixed (r, lane>>4)).
__device__ __forceinline__ void layerNorm(v8f acc[8], const float* __restrict__ g,
                                          const float* __restrict__ beta, int lane) {
    const int lo = lane & 15;
    float s1[8], s2[8];
#pragma unroll
    for (int r = 0; r < 8; ++r) { s1[r] = 0.0f; s2[r] = 0.0f; }
#pragma unroll
    for (int ct = 0; ct < 8; ++ct)
#pragma unroll
        for (int r = 0; r < 8; ++r) { const float v = acc[ct][r]; s1[r] += v; s2[r] += v * v; }
#pragma unroll
    for (int m = 1; m < 16; m <<= 1) {
#pragma unroll
        for (int r = 0; r < 8; ++r) {
            s1[r] += __shfl_xor(s1[r], m, 32);
            s2[r] += __shfl_xor(s2[r], m, 32);
        }
    }
    float mean[8], inv[8];
#pragma unroll
    for (int r = 0; r < 8; ++r) {
        mean[r] = s1[r] * (1.0f / 128.0f);
        const float var = s2[r] * (1.0f / 128.0f) - mean[r] * mean[r];
        inv[r] = rsqrtf(var + LN_EPS);
    }
#pragma unroll
    for (int ct = 0; ct < 8; ++ct) {
        const float gv = g[ct * 16 + lo], bv = beta[ct * 16 + lo];
#pragma unroll
        for (int r = 0; r < 8; ++r)
            acc[ct][r] = (acc[ct][r] - mean[r]) * inv[r] * gv + bv;
    }
}

// ---------------------------------------------------------------- kernels ---

__global__ __launch_bounds__(256) void zero_f(float* __restrict__ p, long long n) {
    const long long i = (long long)blockIdx.x * 256 + threadIdx.x;
    if (i < n) p[i] = 0.0f;
}

// fp32 W [K x N] -> f16 W^T [Npad x Kpad], zero-padded. blockIdx.y = stacked step.
__global__ __launch_bounds__(256) void pack_wt(const float* __restrict__ W, h16* __restrict__ Wt,
                                               int K, int N, int Kpad, int Npad) {
    W  += (long long)blockIdx.y * K * N;
    Wt += (long long)blockIdx.y * Kpad * Npad;
    const int idx = blockIdx.x * 256 + threadIdx.x;
    if (idx >= Kpad * Npad) return;
    const int n = idx / Kpad, k = idx - n * Kpad;
    const float v = (k < K && n < N) ? W[(long long)k * N + n] : 0.0f;
    Wt[idx] = (h16)v;
}

// Encoder: in [R x Din] -> LN(MLP) -> out [R x 128] (+ optional f16 shadow).
__global__ __launch_bounds__(128) void encoder_kernel(
    const float* __restrict__ in, int Din, int R,
    const h16* __restrict__ W1t, const float* __restrict__ b1,
    const h16* __restrict__ W2t, const float* __restrict__ b2,
    const float* __restrict__ g, const float* __restrict__ beta,
    float* __restrict__ out, h16* __restrict__ shadow) {
    constexpr int SA = 40, SH = 136, SF = 132;
    constexpr int WB = 16 * SF * 4;   // 8448B >= A-stage (1280) and hidden (4352)
    __shared__ __align__(16) unsigned char smem[4 * WB];
    const int wave = threadIdx.x >> 5, lane = threadIdx.x & 31;
    h16*   aLds = (h16*)(smem + wave * WB);
    h16*   hLds = (h16*)(smem + wave * WB);
    float* fLds = (float*)(smem + wave * WB);

    const int tile  = blockIdx.x * 4 + wave;
    const bool act  = tile * 16 < R;
    const int rbase = act ? tile * 16 : 0;

    for (int r = 0; r < 16; ++r) {
        int row = rbase + r; if (row >= R) row = R - 1;
        const float v = (lane < Din) ? in[(size_t)row * Din + lane] : 0.0f;
        aLds[r * SA + lane] = (h16)v;
    }
    __builtin_amdgcn_wave_barrier();

    v8f acc[8]; zeroAcc(acc);
    waveGemm(aLds, SA, W1t, 32, acc, lane);
    biasAct(acc, b1, lane, true);
    __builtin_amdgcn_wave_barrier();
    storeHidden(acc, hLds, SH, lane);
    __builtin_amdgcn_wave_barrier();

    v8f o[8]; zeroAcc(o);
    waveGemm(hLds, SH, W2t, 128, o, lane);
    biasAct(o, b2, lane, false);
    layerNorm(o, g, beta, lane);
    __builtin_amdgcn_wave_barrier();
    storeOutF(o, fLds, SF, lane);
    __builtin_amdgcn_wave_barrier();

    if (act) {
        for (int r = 0; r < 16; ++r) {
            const int row = rbase + r;
            if (row < R) {
#pragma unroll
                for (int j = 0; j < 4; ++j) {
                    const int c = lane + j * 32;
                    const float v = fLds[r * SF + c];
                    out[(size_t)row * HDIM + c] = v;
                    if (shadow != nullptr) shadow[(size_t)row * HDIM + c] = (h16)v;
                }
            }
        }
    }
}

// Edge update: e_in = [nf16[snd] | nf16[rcv] | ef] (K=384) -> LN(MLP) = e_up;
// ef += e_up (store-only, fp32 tile parked in LDS); agg[rcv] += e_up (f32 atomics).
// NEDGES % 64 == 0 -> no guards.
__global__ __launch_bounds__(128) void edge_step_kernel(
    const h16* __restrict__ nf16, float* __restrict__ ef, float* __restrict__ agg,
    const int* __restrict__ ei,
    const h16* __restrict__ W1t, const float* __restrict__ b1,
    const h16* __restrict__ W2t, const float* __restrict__ b2,
    const float* __restrict__ g, const float* __restrict__ beta) {
    constexpr int SA = 392, SH = 136, SF = 132;
    constexpr int WB_OV = 16 * SA * 2;   // 12544B overlay: A-stage / hidden / f32-out
    constexpr int WB_EF = 16 * SF * 4;   // 8448B parked fp32 ef tile
    constexpr int WB    = WB_OV + WB_EF; // 20992B per wave
    __shared__ __align__(16) unsigned char smem[4 * WB];
    const int wave = threadIdx.x >> 5, lane = threadIdx.x & 31;
    h16*   aLds  = (h16*)(smem + wave * WB);
    h16*   hLds  = (h16*)(smem + wave * WB);
    float* fLds  = (float*)(smem + wave * WB);
    float* efLds = (float*)(smem + wave * WB + WB_OV);

    const int ebase = (blockIdx.x * 4 + wave) * 16;

    for (int r = 0; r < 16; ++r) {
        const int e = ebase + r;
        const int s = ei[e];
        const int d = ei[NEDGES + e];
        const h16*   ps = nf16 + (size_t)s * HDIM;
        const h16*   pd = nf16 + (size_t)d * HDIM;
        const float* pe = ef + (size_t)e * HDIM;
        // f16 gathers: 2 halves per lane -> 128B/wave requests
#pragma unroll
        for (int j = 0; j < 2; ++j) {
            const int c = lane * 2 + j * 64;
            *(unsigned int*)(aLds + r * SA +       c) = *(const unsigned int*)(ps + c);
            *(unsigned int*)(aLds + r * SA + 128 + c) = *(const unsigned int*)(pd + c);
        }
        // ef: read fp32 once; keep fp32 in LDS for the residual, f16 for WMMA
#pragma unroll
        for (int j = 0; j < 4; ++j) {
            const int c = lane + j * 32;
            const float v = pe[c];
            aLds[r * SA + 256 + c] = (h16)v;
            efLds[r * SF + c] = v;
        }
    }
    __builtin_amdgcn_wave_barrier();

    v8f acc[8]; zeroAcc(acc);
    waveGemm(aLds, SA, W1t, 384, acc, lane);        // 12 K-chunks x 8 tiles = 96 WMMA
    biasAct(acc, b1, lane, true);
    __builtin_amdgcn_wave_barrier();
    storeHidden(acc, hLds, SH, lane);
    __builtin_amdgcn_wave_barrier();

    v8f o[8]; zeroAcc(o);
    waveGemm(hLds, SH, W2t, 128, o, lane);          // 32 WMMA
    biasAct(o, b2, lane, false);
    layerNorm(o, g, beta, lane);
    __builtin_amdgcn_wave_barrier();
    storeOutF(o, fLds, SF, lane);
    __builtin_amdgcn_wave_barrier();

    for (int r = 0; r < 16; ++r) {
        const int e = ebase + r;
        const int d = ei[NEDGES + e];
#pragma unroll
        for (int j = 0; j < 4; ++j) {
            const int c = lane + j * 32;
            const float v = fLds[r * SF + c];
            ef[(size_t)e * HDIM + c] = efLds[r * SF + c] + v;   // residual, store-only
            atomicAdd(&agg[(size_t)d * HDIM + c], v);           // scatter-add (coalesced)
        }
    }
}

// Node update: n_in = [nf | agg] (K=256) -> LN(MLP) = n_up; nf += n_up (+ f16 shadow).
__global__ __launch_bounds__(128) void node_step_kernel(
    float* __restrict__ nf, h16* __restrict__ nf16, const float* __restrict__ agg,
    const h16* __restrict__ W1t, const float* __restrict__ b1,
    const h16* __restrict__ W2t, const float* __restrict__ b2,
    const float* __restrict__ g, const float* __restrict__ beta) {
    constexpr int SA = 264, SH = 136, SF = 132;
    constexpr int WB_OV = 16 * SA * 2;   // 8448B overlay == f32-out size
    constexpr int WB_NF = 16 * SF * 4;   // 8448B parked fp32 nf tile
    constexpr int WB    = WB_OV + WB_NF;
    __shared__ __align__(16) unsigned char smem[4 * WB];
    const int wave = threadIdx.x >> 5, lane = threadIdx.x & 31;
    h16*   aLds  = (h16*)(smem + wave * WB);
    h16*   hLds  = (h16*)(smem + wave * WB);
    float* fLds  = (float*)(smem + wave * WB);
    float* nfLds = (float*)(smem + wave * WB + WB_OV);

    const int tile  = blockIdx.x * 4 + wave;
    const bool act  = tile * 16 < NNODES;
    const int nbase = act ? tile * 16 : 0;

    for (int r = 0; r < 16; ++r) {
        int node = nbase + r; if (node >= NNODES) node = NNODES - 1;
        const float* pn = nf  + (size_t)node * HDIM;
        const float* pa = agg + (size_t)node * HDIM;
#pragma unroll
        for (int j = 0; j < 4; ++j) {
            const int c = lane + j * 32;
            const float v = pn[c];
            aLds[r * SA +       c] = (h16)v;
            nfLds[r * SF + c] = v;
            aLds[r * SA + 128 + c] = (h16)pa[c];
        }
    }
    __builtin_amdgcn_wave_barrier();

    v8f acc[8]; zeroAcc(acc);
    waveGemm(aLds, SA, W1t, 256, acc, lane);
    biasAct(acc, b1, lane, true);
    __builtin_amdgcn_wave_barrier();
    storeHidden(acc, hLds, SH, lane);
    __builtin_amdgcn_wave_barrier();

    v8f o[8]; zeroAcc(o);
    waveGemm(hLds, SH, W2t, 128, o, lane);
    biasAct(o, b2, lane, false);
    layerNorm(o, g, beta, lane);
    __builtin_amdgcn_wave_barrier();
    storeOutF(o, fLds, SF, lane);
    __builtin_amdgcn_wave_barrier();

    if (act) {
        for (int r = 0; r < 16; ++r) {
            const int node = nbase + r;
            if (node < NNODES) {
#pragma unroll
                for (int j = 0; j < 4; ++j) {
                    const int c = lane + j * 32;
                    const float nv = nfLds[r * SF + c] + fLds[r * SF + c];
                    nf[(size_t)node * HDIM + c]   = nv;       // residual, store-only
                    nf16[(size_t)node * HDIM + c] = (h16)nv;  // shadow for gathers
                }
            }
        }
    }
}

// Decoder: nf16 -> relu(nf@W1+b1)@W2+b2 -> out [N x 3]. W2^T padded to [16 x 128].
__global__ __launch_bounds__(128) void decoder_kernel(
    const h16* __restrict__ nf16,
    const h16* __restrict__ W1t, const float* __restrict__ b1,
    const h16* __restrict__ W2t, const float* __restrict__ b2,
    float* __restrict__ out) {
    constexpr int SA = 136, SH = 136;
    constexpr int WB = 16 * SA * 2;   // 4352B
    __shared__ __align__(16) unsigned char smem[4 * WB];
    const int wave = threadIdx.x >> 5, lane = threadIdx.x & 31;
    h16* aLds = (h16*)(smem + wave * WB);
    h16* hLds = (h16*)(smem + wave * WB);

    const int tile  = blockIdx.x * 4 + wave;
    const bool act  = tile * 16 < NNODES;
    const int nbase = act ? tile * 16 : 0;

    for (int r = 0; r < 16; ++r) {
        int node = nbase + r; if (node >= NNODES) node = NNODES - 1;
        const h16* pn = nf16 + (size_t)node * HDIM;
#pragma unroll
        for (int j = 0; j < 2; ++j) {
            const int c = lane * 2 + j * 64;
            *(unsigned int*)(aLds + r * SA + c) = *(const unsigned int*)(pn + c);
        }
    }
    __builtin_amdgcn_wave_barrier();

    v8f acc[8]; zeroAcc(acc);
    waveGemm(aLds, SA, W1t, 128, acc, lane);
    biasAct(acc, b1, lane, true);
    __builtin_amdgcn_wave_barrier();
    storeHidden(acc, hLds, SH, lane);
    __builtin_amdgcn_wave_barrier();

    v8f o;
#pragma unroll
    for (int r = 0; r < 8; ++r) o[r] = 0.0f;
    for (int kc = 0; kc < 128; kc += 32) {
        v16h a = ldsAFrag(hLds, SH, lane, kc);
        v16h b = glbBFrag(W2t, 128, 0, kc, lane);
        o = wmma16(a, b, o);
    }
    const int lo = lane & 15, hi = lane >> 4;
    if (act && lo < 3) {
        const float bb = b2[lo];
#pragma unroll
        for (int r = 0; r < 8; ++r) {
            const int node = nbase + r + 8 * hi;
            if (node < NNODES) out[(size_t)node * 3 + lo] = o[r] + bb;
        }
    }
}

// ------------------------------------------------------------------- host ---

extern "C" void kernel_launch(void* const* d_in, const int* in_sizes, int n_in,
                              void* d_out, int out_size, void* d_ws, size_t ws_size,
                              hipStream_t stream) {
    (void)n_in; (void)out_size; (void)ws_size;
    auto F = [&](int i) { return (const float*)d_in[i]; };

    const float *x, *edge_attr; const int* ei;
    const float *enW1, *enb1, *enW2, *enb2, *eng, *enbt;
    const float *eeW1, *eeb1, *eeW2, *eeb2, *eeg, *eebt;
    const float *eW1, *eb1, *eW2, *eb2, *eg, *ebt;      // stacked [15,...]
    const float *nW1, *nb1, *nW2, *nb2, *ng, *nbt;      // stacked [15,...]
    const float *dW1, *db1, *dW2, *db2;

    // JAX tree_flatten sorts dict keys; detect vs insertion order via sizes.
    if (in_sizes[0] == STEPS * 384 * HDIM) {            // sorted-key order
        eW1 = F(0);  eW2 = F(1);  eb1 = F(2);  eb2 = F(3);  ebt = F(4);  eg = F(5);
        nW1 = F(6);  nW2 = F(7);  nb1 = F(8);  nb2 = F(9);  nbt = F(10); ng = F(11);
        dW1 = F(12); dW2 = F(13); db1 = F(14); db2 = F(15);
        edge_attr = F(16); ei = (const int*)d_in[17];
        eeW1 = F(18); eeW2 = F(19); eeb1 = F(20); eeb2 = F(21); eebt = F(22); eeg = F(23);
        enW1 = F(24); enW2 = F(25); enb1 = F(26); enb2 = F(27); enbt = F(28); eng = F(29);
        x = F(30);
    } else {                                            // insertion order
        x = F(0); edge_attr = F(1); ei = (const int*)d_in[2];
        enW1 = F(3);  enb1 = F(4);  enW2 = F(5);  enb2 = F(6);  eng = F(7);  enbt = F(8);
        eeW1 = F(9);  eeb1 = F(10); eeW2 = F(11); eeb2 = F(12); eeg = F(13); eebt = F(14);
        eW1 = F(15);  eb1 = F(16);  eW2 = F(17);  eb2 = F(18);  eg = F(19);  ebt = F(20);
        nW1 = F(21);  nb1 = F(22);  nW2 = F(23);  nb2 = F(24);  ng = F(25);  nbt = F(26);
        dW1 = F(27);  db1 = F(28);  dW2 = F(29);  db2 = F(30);
    }

    char* ws = (char*)d_ws;
    size_t off = 0;
    auto take = [&](size_t bytes) { char* p = ws + off; off = (off + bytes + 255) & ~(size_t)255; return p; };
    float* nf   = (float*)take((size_t)NNODES * HDIM * 4);
    float* agg  = (float*)take((size_t)NNODES * HDIM * 4);
    float* ef   = (float*)take((size_t)NEDGES * HDIM * 4);
    h16*   nf16 = (h16*)take((size_t)NNODES * HDIM * 2);
    h16* enW1t = (h16*)take(128 * 32 * 2);
    h16* enW2t = (h16*)take(128 * 128 * 2);
    h16* eeW1t = (h16*)take(128 * 32 * 2);
    h16* eeW2t = (h16*)take(128 * 128 * 2);
    h16* eW1t  = (h16*)take((size_t)STEPS * 128 * 384 * 2);
    h16* eW2t  = (h16*)take((size_t)STEPS * 128 * 128 * 2);
    h16* nW1t  = (h16*)take((size_t)STEPS * 128 * 256 * 2);
    h16* nW2t  = (h16*)take((size_t)STEPS * 128 * 128 * 2);
    h16* dW1t  = (h16*)take(128 * 128 * 2);
    h16* dW2t  = (h16*)take(16 * 128 * 2);

    // Pack all weights -> f16 W^T, padded (K to mult. of 32, N to mult. of 16).
    pack_wt<<<dim3(16, 1),      256, 0, stream>>>(enW1, enW1t, 12, 128, 32, 128);
    pack_wt<<<dim3(64, 1),      256, 0, stream>>>(enW2, enW2t, 128, 128, 128, 128);
    pack_wt<<<dim3(16, 1),      256, 0, stream>>>(eeW1, eeW1t, 4, 128, 32, 128);
    pack_wt<<<dim3(64, 1),      256, 0, stream>>>(eeW2, eeW2t, 128, 128, 128, 128);
    pack_wt<<<dim3(192, STEPS), 256, 0, stream>>>(eW1, eW1t, 384, 128, 384, 128);
    pack_wt<<<dim3(64, STEPS),  256, 0, stream>>>(eW2, eW2t, 128, 128, 128, 128);
    pack_wt<<<dim3(128, STEPS), 256, 0, stream>>>(nW1, nW1t, 256, 128, 256, 128);
    pack_wt<<<dim3(64, STEPS),  256, 0, stream>>>(nW2, nW2t, 128, 128, 128, 128);
    pack_wt<<<dim3(64, 1),      256, 0, stream>>>(dW1, dW1t, 128, 128, 128, 128);
    pack_wt<<<dim3(8, 1),       256, 0, stream>>>(dW2, dW2t, 128, 3, 128, 16);

    const int nodeBlocks = (NNODES / 16 + 3) / 4;   // 782 (4 waves x 16-row tiles)
    const int edgeBlocks = NEDGES / 64;             // 9375

    // Encoders (node encoder also fills the f16 shadow used for edge gathers)
    encoder_kernel<<<nodeBlocks, 128, 0, stream>>>(x, 12, NNODES, enW1t, enb1, enW2t, enb2,
                                                   eng, enbt, nf, nf16);
    encoder_kernel<<<edgeBlocks, 128, 0, stream>>>(edge_attr, 4, NEDGES, eeW1t, eeb1, eeW2t, eeb2,
                                                   eeg, eebt, ef, (h16*)nullptr);

    // 15 message-passing steps
    const long long aggN = (long long)NNODES * HDIM;
    for (int s = 0; s < STEPS; ++s) {
        zero_f<<<(unsigned)((aggN + 255) / 256), 256, 0, stream>>>(agg, aggN);
        edge_step_kernel<<<edgeBlocks, 128, 0, stream>>>(
            nf16, ef, agg, ei,
            eW1t + (size_t)s * 128 * 384, eb1 + s * 128,
            eW2t + (size_t)s * 128 * 128, eb2 + s * 128,
            eg + s * 128, ebt + s * 128);
        node_step_kernel<<<nodeBlocks, 128, 0, stream>>>(
            nf, nf16, agg,
            nW1t + (size_t)s * 128 * 256, nb1 + s * 128,
            nW2t + (size_t)s * 128 * 128, nb2 + s * 128,
            ng + s * 128, nbt + s * 128);
    }

    // Decoder -> d_out [NNODES x 3] fp32
    decoder_kernel<<<nodeBlocks, 128, 0, stream>>>(nf16, dW1t, db1, dW2t, db2, (float*)d_out);
}
